// MultiHeadedAttention_65575560675479
// MI455X (gfx1250) — matrix operand order repaired and checked by reading, not compiled, not run on hardware
//
#include <hip/hip_runtime.h>
#include <hip/hip_bf16.h>

// MI455X (gfx1250) multi-head causal attention, bf16 WMMA pipeline.
// B=8, T=2048, C=512, H=8, D=64.  ~103 GFLOP vs ~100MB HBM traffic -> compute
// bound -> all four GEMM stages run on v_wmma_f32_16x16x32_bf16.
// Byte-copy LDS staging (K tile, proj input tile) uses the CDNA5 async
// global->LDS DMA path (ASYNCcnt) instead of VGPR round-trips.
//
// Workspace layout (bf16 stored as ushort):
//   Qb [B,H,T,D]  8,388,608 elems
//   Kb [B,H,T,D]  8,388,608
//   Vb [B,H,T,D]  8,388,608
//   Ob [B,T,H*D]  8,388,608   (concat-head attention output, proj input)
// total 64 MB of d_ws.

#define B_SZ 8
#define T_SZ 2048
#define C_SZ 512
#define H_SZ 8
#define D_SZ 64

typedef __attribute__((ext_vector_type(16))) __bf16 v16bf;
typedef __attribute__((ext_vector_type(8)))  float  v8f;

union FragU {
  v16bf v;
  uint4 q[2];
};

static __device__ __forceinline__ unsigned short f2bf(float f) {
  unsigned int u = __float_as_uint(f);
  unsigned int r = u + 0x7FFFu + ((u >> 16) & 1u);   // round-to-nearest-even
  return (unsigned short)(r >> 16);
}

static __device__ __forceinline__ v8f vzero8() {
  v8f z = {0.f, 0.f, 0.f, 0.f, 0.f, 0.f, 0.f, 0.f};
  return z;
}

static __device__ __forceinline__ v8f wmma_bf16(const FragU& a, const FragU& b, v8f c) {
  return __builtin_amdgcn_wmma_f32_16x16x32_bf16(
      /*neg_a=*/false, a.v, /*neg_b=*/false, b.v,
      /*c_mod=*/(short)0, c, /*reuse_a=*/false, /*reuse_b=*/false);
}

// CDNA5 async global->LDS DMA: copies 16B from global addr to LDS addr,
// tracked by ASYNCcnt.  First operand VGPR holds the LDS byte address
// (generic-address low 32 bits == LDS offset per the aperture encoding).
static __device__ __forceinline__ void async_ld_b128(const void* g, unsigned lds_addr) {
  asm volatile("global_load_async_to_lds_b128 %0, %1, off"
               :: "v"(lds_addr), "v"(g) : "memory");
}
static __device__ __forceinline__ void wait_async0() {
  asm volatile("s_wait_asynccnt 0" ::: "memory");
}

// ---------------------------------------------------------------------------
// Kernel 1: QKV projection.  For each (b,h): Q/K/V[t,d] = sum_c x[b,t,c]*W[h,c,d]
// Tile: 128 rows x 64 cols per block, K-steps of 32.  8 waves, each wave owns a
// 16-row strip and 4 N-tiles, accumulating Q,K,V simultaneously (x tile reuse).
// (f32->bf16 conversion happens on the way into LDS, so no async path here.)
// ---------------------------------------------------------------------------
__global__ __launch_bounds__(256) void qkv_kernel(
    const float* __restrict__ x,
    const float* __restrict__ Wq, const float* __restrict__ Wk,
    const float* __restrict__ Wv,
    unsigned short* __restrict__ Qb, unsigned short* __restrict__ Kb,
    unsigned short* __restrict__ Vb) {
  __shared__ __attribute__((aligned(16))) unsigned short ldsX[128 * 32];     // [row][k]
  __shared__ __attribute__((aligned(16))) unsigned short ldsW[3][64 * 40];   // [n][k] padded

  const int blk = blockIdx.x;
  const int mt = blk & 15;          // T/128 = 16 row tiles
  const int bh = blk >> 4;
  const int h  = bh & (H_SZ - 1);
  const int b  = bh >> 3;
  const int tid  = threadIdx.x;
  const int wave = tid >> 5;
  const int lane = tid & 31;
  const int half = lane >> 4;
  const int lm   = lane & 15;
  const int row0 = mt * 128;

  v8f accQ[4], accK[4], accV[4];
#pragma unroll
  for (int i = 0; i < 4; ++i) { accQ[i] = vzero8(); accK[i] = vzero8(); accV[i] = vzero8(); }

  for (int k0 = 0; k0 < C_SZ; k0 += 32) {
    __syncthreads();
    // stage x tile [128][32] as bf16, row-major
    for (int i = tid; i < 128 * 32; i += 256) {
      int r = i >> 5, c = i & 31;
      ldsX[i] = f2bf(x[((size_t)b * T_SZ + row0 + r) * C_SZ + k0 + c]);
    }
    // stage W tiles transposed: ldsW[m][n*40+k] = W[h, k0+k, n]
    for (int i = tid; i < 64 * 32; i += 256) {
      int n = i & 63, k = i >> 6;
      size_t g = ((size_t)h * C_SZ + (k0 + k)) * D_SZ + n;
      ldsW[0][n * 40 + k] = f2bf(Wq[g]);
      ldsW[1][n * 40 + k] = f2bf(Wk[g]);
      ldsW[2][n * 40 + k] = f2bf(Wv[g]);
    }
    __syncthreads();

    // A fragment (16x32): lane row = wave*16+lm, k = 16*half+e
    FragU a;
    const int mrow = wave * 16 + lm;
    a.q[0] = *(const uint4*)&ldsX[mrow * 32 + half * 16];
    a.q[1] = *(const uint4*)&ldsX[mrow * 32 + half * 16 + 8];

#pragma unroll
    for (int nt = 0; nt < 4; ++nt) {
      const int nb = nt * 16 + lm;
      FragU bq, bk, bv;
      bq.q[0] = *(const uint4*)&ldsW[0][nb * 40 + half * 16];
      bq.q[1] = *(const uint4*)&ldsW[0][nb * 40 + half * 16 + 8];
      bk.q[0] = *(const uint4*)&ldsW[1][nb * 40 + half * 16];
      bk.q[1] = *(const uint4*)&ldsW[1][nb * 40 + half * 16 + 8];
      bv.q[0] = *(const uint4*)&ldsW[2][nb * 40 + half * 16];
      bv.q[1] = *(const uint4*)&ldsW[2][nb * 40 + half * 16 + 8];
      accQ[nt] = wmma_bf16(a, bq, accQ[nt]);
      accK[nt] = wmma_bf16(a, bk, accK[nt]);
      accV[nt] = wmma_bf16(a, bv, accV[nt]);
    }
  }

  // write back bf16 [B,H,T,D]; C layout: VGPR r -> row (half? 8+r : r), col lm
  const size_t base = ((size_t)(b * H_SZ + h) * T_SZ + row0 + wave * 16);
#pragma unroll
  for (int nt = 0; nt < 4; ++nt) {
#pragma unroll
    for (int r = 0; r < 8; ++r) {
      int rr = half ? (r + 8) : r;
      size_t g = (base + rr) * D_SZ + nt * 16 + lm;
      Qb[g] = f2bf(accQ[nt][r]);
      Kb[g] = f2bf(accK[nt][r]);
      Vb[g] = f2bf(accV[nt][r]);
    }
  }
}

// ---------------------------------------------------------------------------
// Kernel 2: causal flash attention per (b,h).  Block = 128 query rows, wave
// owns 16 rows.  Key loop in tiles of 32 with online softmax (scale=D^-0.5).
// K tile staged by async global->LDS DMA; V tile transposed through VGPRs.
// ---------------------------------------------------------------------------
__global__ __launch_bounds__(256) void attn_kernel(
    const unsigned short* __restrict__ Qb, const unsigned short* __restrict__ Kb,
    const unsigned short* __restrict__ Vb, unsigned short* __restrict__ Ob) {
  __shared__ __attribute__((aligned(16))) unsigned short ldsK[32 * 72];   // [s][d] padded
  __shared__ __attribute__((aligned(16))) unsigned short ldsVt[64 * 40];  // [d][s] padded
  __shared__ __attribute__((aligned(16))) unsigned short ldsP[8][16 * 32]; // per-wave P patch

  const int blk = blockIdx.x;
  const int mt = blk & 15;
  const int bh = blk >> 4;
  const int h  = bh & (H_SZ - 1);
  const int b  = bh >> 3;
  const int tid  = threadIdx.x;
  const int wave = tid >> 5;
  const int lane = tid & 31;
  const int half = lane >> 4;
  const int lm   = lane & 15;
  const int q0   = mt * 128 + wave * 16;   // wave's first query row

  // Q fragments for this wave's 16 rows, 2 K-steps of 32 over D=64
  const unsigned short* Qrow =
      Qb + ((size_t)(b * H_SZ + h) * T_SZ + q0 + lm) * D_SZ;
  FragU qa[2];
  qa[0].q[0] = *(const uint4*)(Qrow + half * 16);
  qa[0].q[1] = *(const uint4*)(Qrow + half * 16 + 8);
  qa[1].q[0] = *(const uint4*)(Qrow + 32 + half * 16);
  qa[1].q[1] = *(const uint4*)(Qrow + 32 + half * 16 + 8);

  v8f o[4];
#pragma unroll
  for (int i = 0; i < 4; ++i) o[i] = vzero8();
  float mrun[8], lrun[8];
#pragma unroll
  for (int r = 0; r < 8; ++r) { mrun[r] = -1e30f; lrun[r] = 0.f; }

  const float scale = 0.125f;  // 1/sqrt(64)
  const int nkt = (mt + 1) * 4;  // causal: key tiles of 32 up to block's last row

  const unsigned short* Kg0 = Kb + (size_t)(b * H_SZ + h) * T_SZ * D_SZ;
  const unsigned short* Vg0 = Vb + (size_t)(b * H_SZ + h) * T_SZ * D_SZ;

  // per-thread 16B chunk of the 32x64 bf16 tile (256 chunks == 256 threads)
  const int cs   = tid >> 3;         // key row within tile
  const int coff = (tid & 7) * 8;    // d offset (8 ushorts = 16B)

  for (int kt = 0; kt < nkt; ++kt) {
    const int s0 = kt * 32;
    __syncthreads();

    // K tile: async DMA straight into padded LDS (byte copy, stride 144B)
    async_ld_b128(Kg0 + (size_t)(s0 + cs) * D_SZ + coff,
                  (unsigned)(uintptr_t)&ldsK[cs * 72 + coff]);

    // V tile: one b128 global load per thread, transpose-scatter into LDS
    {
      const uint4 vv = *(const uint4*)(Vg0 + (size_t)(s0 + cs) * D_SZ + coff);
      const unsigned short* pv = (const unsigned short*)&vv;
#pragma unroll
      for (int e = 0; e < 8; ++e)
        ldsVt[(coff + e) * 40 + cs] = pv[e];
    }

    // prefetch next tile while this one computes
    if (kt + 1 < nkt) {
      __builtin_prefetch(Kg0 + (size_t)(s0 + 32 + cs) * D_SZ + coff, 0, 0);
      __builtin_prefetch(Vg0 + (size_t)(s0 + 32 + cs) * D_SZ + coff, 0, 0);
    }

    wait_async0();
    __syncthreads();

    // scores: two 16x16 subtiles over 32 keys; B(k=d, n=s) = K[s,d]
    v8f sc[2];
#pragma unroll
    for (int j = 0; j < 2; ++j) {
      const int n = j * 16 + lm;
      FragU b0, b1;
      b0.q[0] = *(const uint4*)&ldsK[n * 72 + half * 16];
      b0.q[1] = *(const uint4*)&ldsK[n * 72 + half * 16 + 8];
      b1.q[0] = *(const uint4*)&ldsK[n * 72 + 32 + half * 16];
      b1.q[1] = *(const uint4*)&ldsK[n * 72 + 32 + half * 16 + 8];
      v8f z = vzero8();
      z = wmma_bf16(qa[0], b0, z);
      z = wmma_bf16(qa[1], b1, z);
      sc[j] = z;
    }

    // online softmax; row of slot r lives in 16 lanes of this half
#pragma unroll
    for (int r = 0; r < 8; ++r) {
      const int rowg = q0 + (half ? (r + 8) : r);
      float x0 = sc[0][r] * scale;
      float x1 = sc[1][r] * scale;
      if (s0 + lm > rowg)      x0 = -1e30f;
      if (s0 + 16 + lm > rowg) x1 = -1e30f;
      float mx = fmaxf(x0, x1);
#pragma unroll
      for (int off = 1; off < 16; off <<= 1)
        mx = fmaxf(mx, __shfl_xor(mx, off, 32));
      mx = fmaxf(mx, mrun[r]);
      const float alpha = __expf(mrun[r] - mx);
      const float p0 = __expf(x0 - mx);
      const float p1 = __expf(x1 - mx);
      float ps = p0 + p1;
#pragma unroll
      for (int off = 1; off < 16; off <<= 1)
        ps += __shfl_xor(ps, off, 32);
      lrun[r] = lrun[r] * alpha + ps;
      mrun[r] = mx;
#pragma unroll
      for (int nt = 0; nt < 4; ++nt) o[nt][r] *= alpha;
      const int prow = half ? (r + 8) : r;
      ldsP[wave][prow * 32 + lm]      = f2bf(p0);
      ldsP[wave][prow * 32 + 16 + lm] = f2bf(p1);
    }
    // wave-local LDS handoff (per-wave region; lockstep wave32, DS in-order)
    asm volatile("s_wait_dscnt 0" ::: "memory");
    __builtin_amdgcn_wave_barrier();

    // A fragment of P (16x32), then O += P x V
    FragU pa;
    pa.q[0] = *(const uint4*)&ldsP[wave][lm * 32 + half * 16];
    pa.q[1] = *(const uint4*)&ldsP[wave][lm * 32 + half * 16 + 8];
#pragma unroll
    for (int nt = 0; nt < 4; ++nt) {
      const int n = nt * 16 + lm;
      FragU vb;
      vb.q[0] = *(const uint4*)&ldsVt[n * 40 + half * 16];
      vb.q[1] = *(const uint4*)&ldsVt[n * 40 + half * 16 + 8];
      o[nt] = wmma_bf16(pa, vb, o[nt]);
    }
  }

  // epilogue: normalize and write concat-head bf16 output [B,T,H*D]
#pragma unroll
  for (int nt = 0; nt < 4; ++nt) {
#pragma unroll
    for (int r = 0; r < 8; ++r) {
      const int rowg = q0 + (half ? (r + 8) : r);
      const float val = o[nt][r] / lrun[r];
      Ob[((size_t)b * T_SZ + rowg) * (H_SZ * D_SZ) + h * D_SZ + nt * 16 + lm] =
          f2bf(val);
    }
  }
}

// ---------------------------------------------------------------------------
// Kernel 3: output projection.  [16384 x 512] @ Wproj[512 x 512] + bias -> f32
// Ob tile (already bf16, byte copy) staged via async global->LDS DMA.
// ---------------------------------------------------------------------------
__global__ __launch_bounds__(256) void proj_kernel(
    const unsigned short* __restrict__ Ob, const float* __restrict__ Wp,
    const float* __restrict__ bias, float* __restrict__ out) {
  __shared__ __attribute__((aligned(16))) unsigned short ldsO[128 * 32];
  __shared__ __attribute__((aligned(16))) unsigned short ldsW[64 * 40];

  const int blk = blockIdx.x;
  const int ntile8 = blk & 7;     // 512/64 = 8 col tiles
  const int mtile  = blk >> 3;    // 16384/128 = 128 row tiles
  const int row0 = mtile * 128;
  const int n0   = ntile8 * 64;
  const int tid  = threadIdx.x;
  const int wave = tid >> 5;
  const int lane = tid & 31;
  const int half = lane >> 4;
  const int lm   = lane & 15;

  v8f acc[4];
#pragma unroll
  for (int i = 0; i < 4; ++i) acc[i] = vzero8();

  for (int k0 = 0; k0 < 512; k0 += 32) {
    __syncthreads();
    // O tile [128][32] bf16: 512 16B chunks, async DMA, 2 per thread
#pragma unroll
    for (int j = 0; j < 2; ++j) {
      int chunk = tid + j * 256;
      int r = chunk >> 2, c8 = (chunk & 3) * 8;
      async_ld_b128(Ob + ((size_t)row0 + r) * 512 + k0 + c8,
                    (unsigned)(uintptr_t)&ldsO[r * 32 + c8]);
    }
    // W tile transposed + converted f32->bf16 (VALU path)
    for (int i = tid; i < 64 * 32; i += 256) {
      int n = i & 63, k = i >> 6;
      ldsW[n * 40 + k] = f2bf(Wp[(size_t)(k0 + k) * 512 + n0 + n]);
    }
    wait_async0();
    __syncthreads();

    FragU a;
    const int mrow = wave * 16 + lm;
    a.q[0] = *(const uint4*)&ldsO[mrow * 32 + half * 16];
    a.q[1] = *(const uint4*)&ldsO[mrow * 32 + half * 16 + 8];
#pragma unroll
    for (int nt = 0; nt < 4; ++nt) {
      const int nb = nt * 16 + lm;
      FragU bw;
      bw.q[0] = *(const uint4*)&ldsW[nb * 40 + half * 16];
      bw.q[1] = *(const uint4*)&ldsW[nb * 40 + half * 16 + 8];
      acc[nt] = wmma_bf16(a, bw, acc[nt]);
    }
  }

#pragma unroll
  for (int nt = 0; nt < 4; ++nt) {
    const float bb = bias[n0 + nt * 16 + lm];
#pragma unroll
    for (int r = 0; r < 8; ++r) {
      const int rr = half ? (r + 8) : r;
      out[((size_t)row0 + wave * 16 + rr) * 512 + n0 + nt * 16 + lm] =
          acc[nt][r] + bb;
    }
  }
}

extern "C" void kernel_launch(void* const* d_in, const int* in_sizes, int n_in,
                              void* d_out, int out_size, void* d_ws, size_t ws_size,
                              hipStream_t stream) {
  const float* x     = (const float*)d_in[0];
  const float* Wq    = (const float*)d_in[1];
  const float* Wk    = (const float*)d_in[2];
  const float* Wv    = (const float*)d_in[3];
  const float* Wproj = (const float*)d_in[4];
  const float* bproj = (const float*)d_in[5];
  float* out = (float*)d_out;

  const size_t qkv_elems = (size_t)B_SZ * H_SZ * T_SZ * D_SZ;  // 8,388,608
  unsigned short* Qb = (unsigned short*)d_ws;
  unsigned short* Kb = Qb + qkv_elems;
  unsigned short* Vb = Kb + qkv_elems;
  unsigned short* Ob = Vb + qkv_elems;

  const int blocks = B_SZ * H_SZ * (T_SZ / 128);  // 1024
  qkv_kernel<<<blocks, 256, 0, stream>>>(x, Wq, Wk, Wv, Qb, Kb, Vb);
  attn_kernel<<<blocks, 256, 0, stream>>>(Qb, Kb, Vb, Ob);
  const int pblocks = (B_SZ * T_SZ / 128) * (512 / 64);  // 1024
  proj_kernel<<<pblocks, 256, 0, stream>>>(Ob, Wproj, bproj, out);
}
